// Omni_85152021610756
// MI455X (gfx1250) — matrix operand order repaired and checked by reading, not compiled, not run on hardware
//
#include <hip/hip_runtime.h>
#include <hip/hip_bf16.h>

// ---------------- types ----------------
typedef __bf16 bf16;
typedef __attribute__((ext_vector_type(16))) __bf16 v16bf;
typedef __attribute__((ext_vector_type(8)))  float  v8f;
typedef __attribute__((ext_vector_type(4)))  unsigned int v4u;

// ---------------- problem sizes ----------------
#define BATCH 128
#define SEQ   512
#define NF    64
#define HID   1024
#define ZD    16
#define G3H   3072         // 3*HID
#define LOG2PI 1.8378770664093453f

#define NBLK  128          // persistent grid blocks
#define NTHR  256          // 8 waves per block -> 1024 waves total
#define LDA   1032         // padded LDS row stride (elements) to avoid bank conflicts

// ---------------- workspace layout (bytes, all 256B aligned) ----------------
constexpr size_t OFF_PHI_WHH = 0;                                         // [3072][1024] bf16
constexpr size_t OFF_PHI_WIH = OFF_PHI_WHH + (size_t)G3H*HID*2;           // [3072][64]   bf16
constexpr size_t OFF_TH_WHH  = OFF_PHI_WIH + (size_t)G3H*NF*2;            // [3072][1024] bf16
constexpr size_t OFF_TH_WIH  = OFF_TH_WHH  + (size_t)G3H*HID*2;           // [3072][32]   bf16 (K padded 16->32)
constexpr size_t OFF_XLOC    = OFF_TH_WIH  + (size_t)G3H*32*2;            // [64][1024]   bf16
constexpr size_t OFF_XSCALE  = OFF_XLOC    + (size_t)NF*HID*2;            // [64][1024]   bf16
constexpr size_t OFF_XBF     = OFF_XSCALE  + (size_t)NF*HID*2;            // [512][128][64] bf16 (t-major)
constexpr size_t OFF_HPHI_F  = OFF_XBF     + (size_t)SEQ*BATCH*NF*2;      // 2x [128][1024] f32 (ping-pong)
constexpr size_t OFF_HPHI_B  = OFF_HPHI_F  + (size_t)2*BATCH*HID*4;       // 2x [128][1024] bf16
constexpr size_t OFF_HTH_F   = OFF_HPHI_B  + (size_t)2*BATCH*HID*2;       // 2x [128][1024] f32
constexpr size_t OFF_HTH_B   = OFF_HTH_F   + (size_t)2*BATCH*HID*4;       // 2x [128][1024] bf16
constexpr size_t OFF_Z       = OFF_HTH_B   + (size_t)2*BATCH*HID*2;       // 2x [128][16] f32
constexpr size_t OFF_ZTBF    = OFF_Z       + (size_t)2*BATCH*ZD*4;        // [128][32] bf16 (K padded)
constexpr size_t OFF_RED     = OFF_ZTBF    + (size_t)BATCH*32*2;          // 1024 recon + 2048 kld partials f32
constexpr size_t OFF_BAR     = OFF_RED     + (size_t)3072*4;              // barrier cnt/gen

// ---------------- small math helpers ----------------
__device__ __forceinline__ float sigmoidf_(float x) { return 1.0f / (1.0f + __expf(-x)); }
__device__ __forceinline__ float tanhf_(float x) {
    float xc = fminf(fmaxf(x, -15.0f), 15.0f);
    float e = __expf(2.0f * xc);
    return (e - 1.0f) / (e + 1.0f);
}
__device__ __forceinline__ float softplusf_(float x) {
    return (x > 15.0f) ? x : __logf(1.0f + __expf(x));
}

// ---------------- WMMA helpers (layouts per CDNA5 ISA 7.12.2) ----------------
__device__ __forceinline__ v8f wmma_bf16(v16bf a, v16bf b, v8f c) {
    return __builtin_amdgcn_wmma_f32_16x16x32_bf16(false, a, false, b, (short)0, c, false, false);
}

// A tile 16x32 (MxK), 16-bit, row-major GLOBAL source with leading dim ld.
__device__ __forceinline__ v16bf load_a_tile(const bf16* base, int ld, int m0, int k0, int lane) {
    int row = m0 + (lane & 15);
    int kb  = k0 + ((lane & 16) >> 1);    // +8 for upper half-wave
    const bf16* p = base + (size_t)row * ld + kb;
    union { v16bf v; v4u u[2]; } r;
    r.u[0] = *(const v4u*)(p);
    r.u[1] = *(const v4u*)(p + 16);
    return r.v;
}

// A tile from LDS-staged [16][LDA] bf16
__device__ __forceinline__ v16bf lds_a_tile(const bf16* sA, int k0, int lane) {
    int row = lane & 15;
    int kb  = k0 + ((lane & 16) >> 1);
    const bf16* p = sA + row * LDA + kb;
    union { v16bf v; v4u u[2]; } r;
    r.u[0] = *(const v4u*)(p);
    r.u[1] = *(const v4u*)(p + 16);
    return r.v;
}

// B tile 32x16 (KxN) where B[k][n] = W[n0+n][k], W row-major bf16 with leading dim ld.
__device__ __forceinline__ v16bf load_b_tile(const bf16* W, int ld, int n0, int k0, int lane) {
    int row = n0 + (lane & 15);
    int kb  = k0 + ((lane & 16) >> 1);
    const bf16* p = W + (size_t)row * ld + kb;
    union { v16bf v; v4u u[2]; } r;
    r.u[0] = *(const v4u*)(p);
    r.u[1] = *(const v4u*)(p + 16);
    return r.v;
}

// ---------------- device-wide barrier (sense via generation counter) ----------------
__device__ __forceinline__ void grid_barrier(unsigned* bar) {
    __syncthreads();
    if (threadIdx.x == 0) {
        unsigned* cnt = bar;
        unsigned* gen = bar + 1;
        unsigned g = __hip_atomic_load(gen, __ATOMIC_RELAXED, __HIP_MEMORY_SCOPE_AGENT);
        __threadfence();
        unsigned prev = __hip_atomic_fetch_add(cnt, 1u, __ATOMIC_ACQ_REL, __HIP_MEMORY_SCOPE_AGENT);
        if (prev == (unsigned)(gridDim.x - 1)) {
            __hip_atomic_store(cnt, 0u, __ATOMIC_RELAXED, __HIP_MEMORY_SCOPE_AGENT);
            __hip_atomic_fetch_add(gen, 1u, __ATOMIC_RELEASE, __HIP_MEMORY_SCOPE_AGENT);
        } else {
            while (__hip_atomic_load(gen, __ATOMIC_ACQUIRE, __HIP_MEMORY_SCOPE_AGENT) == g)
                __builtin_amdgcn_s_sleep(2);
        }
    }
    __syncthreads();
}

// ---------------- fused GRU phase: A staged in LDS, K split 2-way per tile ----------------
// Block handles 4 output tiles (same 16 batch rows, 4 jTiles); each tile uses a wave pair:
// half 0: K=[0,512) of Whh + the whole input GEMM + epilogue; half 1: K=[512,1024), LDS handoff.
__device__ __forceinline__ void gru_phase(
    const bf16* __restrict__ hbf_prev, const float* __restrict__ hf_prev,
    const bf16* __restrict__ xin_bf, int Kin,
    const bf16* __restrict__ Whh, const bf16* __restrict__ Wih,
    const float* __restrict__ bih, const float* __restrict__ bhh,
    float* __restrict__ hf_cur, bf16* __restrict__ hbf_cur,
    bf16* sA, float* sRed, int tid)
{
    const int lane = tid & 31;
    const int wib  = tid >> 5;            // wave in block: 0..7
    const int tl   = wib >> 1;            // tile local:    0..3
    const int half = wib & 1;             // K half
    const int tile = blockIdx.x * 4 + tl; // 0..511
    const int m0   = (tile >> 6) << 4;    // same for all waves in block
    const int j0   = (tile & 63) << 4;
    const int n    = lane & 15;
    const int j    = j0 + n;

    // stage A tile [16][HID] bf16 -> LDS [16][LDA] (cooperative, 16B chunks)
    {
        const bf16* src = hbf_prev + (size_t)m0 * HID;
        for (int c = tid; c < (16 * HID) / 8; c += NTHR) {
            int row = c >> 7;
            int col = (c & 127) << 3;
            *(v4u*)(sA + row * LDA + col) = *(const v4u*)(src + (size_t)row * HID + col);
        }
    }
    __syncthreads();

    v8f ar, az, ai, ah;
    if (half == 0) {
        float br = bih[j] + bhh[j];
        float bz = bih[HID + j] + bhh[HID + j];
        float bi = bih[2 * HID + j];
        float bh = bhh[2 * HID + j];
        for (int r = 0; r < 8; ++r) { ar[r] = br; az[r] = bz; ai[r] = bi; ah[r] = bh; }
    } else {
        for (int r = 0; r < 8; ++r) { ar[r] = 0.0f; az[r] = 0.0f; ai[r] = 0.0f; ah[r] = 0.0f; }
    }

    // hidden-to-hidden (this wave's K half): r, z, hn gates
    const int kbeg = half * (HID / 2);
    const int kend = kbeg + (HID / 2);
    for (int k0 = kbeg; k0 < kend; k0 += 32) {
        v16bf a  = lds_a_tile(sA, k0, lane);
        v16bf b0 = load_b_tile(Whh, HID, j0,           k0, lane);
        v16bf b1 = load_b_tile(Whh, HID, HID + j0,     k0, lane);
        v16bf b2 = load_b_tile(Whh, HID, 2 * HID + j0, k0, lane);
        ar = wmma_bf16(a, b0, ar);
        az = wmma_bf16(a, b1, az);
        ah = wmma_bf16(a, b2, ah);
    }

    if (half == 0) {
        // input-to-hidden: r, z, in gates
        for (int k0 = 0; k0 < Kin; k0 += 32) {
            v16bf a  = load_a_tile(xin_bf, Kin, m0, k0, lane);
            v16bf b0 = load_b_tile(Wih, Kin, j0,           k0, lane);
            v16bf b1 = load_b_tile(Wih, Kin, HID + j0,     k0, lane);
            v16bf b2 = load_b_tile(Wih, Kin, 2 * HID + j0, k0, lane);
            ar = wmma_bf16(a, b0, ar);
            az = wmma_bf16(a, b1, az);
            ai = wmma_bf16(a, b2, ai);
        }
    } else {
        // hand partial r/z/hn accumulators to the partner wave
        float* dst = sRed + tl * 768 + (lane << 3);
        *(v4u*)(dst)       = ((const v4u*)&ar)[0];
        *(v4u*)(dst + 4)   = ((const v4u*)&ar)[1];
        *(v4u*)(dst + 256) = ((const v4u*)&az)[0];
        *(v4u*)(dst + 260) = ((const v4u*)&az)[1];
        *(v4u*)(dst + 512) = ((const v4u*)&ah)[0];
        *(v4u*)(dst + 516) = ((const v4u*)&ah)[1];
    }
    __syncthreads();

    if (half == 0) {
        const float* srcR = sRed + tl * 768;
        const int mo = m0 + ((lane >> 4) << 3);
        for (int r = 0; r < 8; ++r) {
            int e = (lane << 3) + r;
            float arr = ar[r] + srcR[e];
            float azz = az[r] + srcR[256 + e];
            float ahh = ah[r] + srcR[512 + e];
            int brow = mo + r;
            float hold = hf_prev[(size_t)brow * HID + j];
            float rg = sigmoidf_(arr);
            float zg = sigmoidf_(azz);
            float ng = tanhf_(ai[r] + rg * ahh);
            float hn = (1.0f - zg) * ng + zg * hold;
            hf_cur[(size_t)brow * HID + j]  = hn;
            hbf_cur[(size_t)brow * HID + j] = (bf16)hn;
        }
    }
}

// ---------------- prepass: convert weights/x to bf16, zero state ----------------
__global__ void vrnn_prep(
    const float* __restrict__ x,
    const float* __restrict__ phi_Wih, const float* __restrict__ phi_Whh,
    const float* __restrict__ th_Wih,  const float* __restrict__ th_Whh,
    const float* __restrict__ x_loc_W, const float* __restrict__ x_scale_W,
    char* __restrict__ ws)
{
    size_t i0 = (size_t)blockIdx.x * blockDim.x + threadIdx.x;
    size_t st = (size_t)gridDim.x * blockDim.x;

    bf16* d;
    d = (bf16*)(ws + OFF_PHI_WHH);
    for (size_t i = i0; i < (size_t)G3H * HID; i += st) d[i] = (bf16)phi_Whh[i];
    d = (bf16*)(ws + OFF_PHI_WIH);
    for (size_t i = i0; i < (size_t)G3H * NF; i += st) d[i] = (bf16)phi_Wih[i];
    d = (bf16*)(ws + OFF_TH_WHH);
    for (size_t i = i0; i < (size_t)G3H * HID; i += st) d[i] = (bf16)th_Whh[i];
    d = (bf16*)(ws + OFF_TH_WIH);                       // pad K 16 -> 32
    for (size_t i = i0; i < (size_t)G3H * 32; i += st) {
        int r = (int)(i >> 5), k = (int)(i & 31);
        d[i] = (k < ZD) ? (bf16)th_Wih[(size_t)r * ZD + k] : (bf16)0.0f;
    }
    d = (bf16*)(ws + OFF_XLOC);
    for (size_t i = i0; i < (size_t)NF * HID; i += st) d[i] = (bf16)x_loc_W[i];
    d = (bf16*)(ws + OFF_XSCALE);
    for (size_t i = i0; i < (size_t)NF * HID; i += st) d[i] = (bf16)x_scale_W[i];

    // x [b][t][f] -> bf16 [t][b][f]
    d = (bf16*)(ws + OFF_XBF);
    for (size_t i = i0; i < (size_t)SEQ * BATCH * NF; i += st) {
        int f = (int)(i % NF);
        size_t r = i / NF;
        int b = (int)(r % BATCH);
        int t = (int)(r / BATCH);
        d[i] = (bf16)x[((size_t)b * SEQ + t) * NF + f];
    }

    // zero initial states (ping-pong buffer 0 only; buffer 1 is written before read)
    float* hf;
    hf = (float*)(ws + OFF_HPHI_F);
    for (size_t i = i0; i < (size_t)BATCH * HID; i += st) hf[i] = 0.0f;
    hf = (float*)(ws + OFF_HTH_F);
    for (size_t i = i0; i < (size_t)BATCH * HID; i += st) hf[i] = 0.0f;
    d = (bf16*)(ws + OFF_HPHI_B);
    for (size_t i = i0; i < (size_t)BATCH * HID; i += st) d[i] = (bf16)0.0f;
    d = (bf16*)(ws + OFF_HTH_B);
    for (size_t i = i0; i < (size_t)BATCH * HID; i += st) d[i] = (bf16)0.0f;
    hf = (float*)(ws + OFF_Z);
    for (size_t i = i0; i < (size_t)BATCH * ZD; i += st) hf[i] = 0.0f;
    d = (bf16*)(ws + OFF_ZTBF);                         // zero incl. K padding
    for (size_t i = i0; i < (size_t)BATCH * 32; i += st) d[i] = (bf16)0.0f;

    if (i0 == 0) {
        unsigned* bar = (unsigned*)(ws + OFF_BAR);
        bar[0] = 0u; bar[1] = 0u;
    }
}

// ---------------- persistent VRNN scan ----------------
__global__ __launch_bounds__(NTHR) void vrnn_persist(
    const float* __restrict__ x, const float* __restrict__ eps,
    const float* __restrict__ phi_bih, const float* __restrict__ phi_bhh,
    const float* __restrict__ phi_loc_W, const float* __restrict__ phi_loc_b,
    const float* __restrict__ phi_scale_W, const float* __restrict__ phi_scale_b,
    const float* __restrict__ th_bih, const float* __restrict__ th_bhh,
    const float* __restrict__ x_loc_b, const float* __restrict__ x_scale_b,
    char* __restrict__ ws, float* __restrict__ out)
{
    const bf16*  phiWhh = (const bf16*)(ws + OFF_PHI_WHH);
    const bf16*  phiWih = (const bf16*)(ws + OFF_PHI_WIH);
    const bf16*  thWhh  = (const bf16*)(ws + OFF_TH_WHH);
    const bf16*  thWih  = (const bf16*)(ws + OFF_TH_WIH);
    const bf16*  xlocW  = (const bf16*)(ws + OFF_XLOC);
    const bf16*  xsclW  = (const bf16*)(ws + OFF_XSCALE);
    const bf16*  xbf    = (const bf16*)(ws + OFF_XBF);
    float*       hphif  = (float*)(ws + OFF_HPHI_F);
    bf16*        hphib  = (bf16*) (ws + OFF_HPHI_B);
    float*       hthf   = (float*)(ws + OFF_HTH_F);
    bf16*        hthb   = (bf16*) (ws + OFF_HTH_B);
    float*       zbuf   = (float*)(ws + OFF_Z);
    bf16*        ztbf   = (bf16*) (ws + OFF_ZTBF);
    float*       red    = (float*)(ws + OFF_RED);       // [0..1023] recon, [1024..3071] kld
    unsigned*    bar    = (unsigned*)(ws + OFF_BAR);

    __shared__ __align__(16) bf16  sA[16 * LDA];        // 33 KB staged A tile
    __shared__ __align__(16) float sRed[4096];          // 16 KB cross-wave reduction

    const int tid  = threadIdx.x;
    const int gtid = blockIdx.x * NTHR + tid;
    const int lane = tid & 31;

    float my_recon = 0.0f;
    float my_kld   = 0.0f;

    for (int t = 0; t < SEQ; ++t) {
        const int pp = t & 1;       // previous-state buffer
        const int pc = pp ^ 1;      // current-state buffer

        // ---- P1: phi GRU (gates GEMM + update), 1024 waves, K split 2-way ----
        gru_phase(hphib + (size_t)pp * BATCH * HID, hphif + (size_t)pp * BATCH * HID,
                  xbf + (size_t)t * BATCH * NF, NF,
                  phiWhh, phiWih, phi_bih, phi_bhh,
                  hphif + (size_t)pc * BATCH * HID, hphib + (size_t)pc * BATCH * HID,
                  sA, sRed, tid);
        grid_barrier(bar);

        // ---- P2: z projection + rsample + KL (f32, 16 lanes per output) ----
        {
            const int idx = gtid >> 4;          // 0..2047  -> (b, i)
            const int sub = gtid & 15;
            const int b = idx >> 4, i = idx & 15;
            const float* hc = hphif + (size_t)pc * BATCH * HID + (size_t)b * HID;
            const float* wL = phi_loc_W   + (size_t)i * (HID + ZD);
            const float* wS = phi_scale_W + (size_t)i * (HID + ZD);
            float aL = 0.0f, aS = 0.0f;
            const int kb = sub * 64;
            for (int k = kb; k < kb + 64; ++k) {
                float hv = hc[k];
                aL += wL[k] * hv;
                aS += wS[k] * hv;
            }
            const float* zp = zbuf + (size_t)pp * BATCH * ZD + (size_t)b * ZD;
            if (sub == 0) {
                for (int k = 0; k < ZD; ++k) {
                    float zv = zp[k];
                    aL += wL[HID + k] * zv;
                    aS += wS[HID + k] * zv;
                }
                aL += phi_loc_b[i];
                aS += phi_scale_b[i];
            }
            aL += __shfl_xor(aL, 1); aL += __shfl_xor(aL, 2);
            aL += __shfl_xor(aL, 4); aL += __shfl_xor(aL, 8);
            aS += __shfl_xor(aS, 1); aS += __shfl_xor(aS, 2);
            aS += __shfl_xor(aS, 4); aS += __shfl_xor(aS, 8);
            if (sub == 0) {
                float zs = softplusf_(aS);
                float e  = eps[(size_t)t * BATCH * ZD + (size_t)b * ZD + i];
                float zn = aL + zs * e;
                zbuf[(size_t)pc * BATCH * ZD + (size_t)b * ZD + i] = zn;
                ztbf[b * 32 + i] = (bf16)zn;
                float dd = aL - zp[i];                         // prior_loc = z_prev
                my_kld += -__logf(zs) + 0.5f * (zs * zs + dd * dd) - 0.5f;
            }
        }
        grid_barrier(bar);

        // ---- P3: theta GRU (input = z, K padded to 32), K split 2-way ----
        gru_phase(hthb + (size_t)pp * BATCH * HID, hthf + (size_t)pp * BATCH * HID,
                  ztbf, 32,
                  thWhh, thWih, th_bih, th_bhh,
                  hthf + (size_t)pc * BATCH * HID, hthb + (size_t)pc * BATCH * HID,
                  sA, sRed, tid);
        grid_barrier(bar);

        // ---- P4: y projections (WMMA), K split 8-way across a block ----
        {
            const int wib = tid >> 5;
            if (blockIdx.x < 32) {
                const int m0 = ((blockIdx.x >> 2) << 4);
                const int f0 = ((blockIdx.x & 3) << 4);
                v8f cl, cs;
                for (int r = 0; r < 8; ++r) { cl[r] = 0.0f; cs[r] = 0.0f; }
                const bf16* hb = hthb + (size_t)pc * BATCH * HID;
                const int kb = wib * (HID / 8);
                for (int k0 = kb; k0 < kb + (HID / 8); k0 += 32) {
                    v16bf a  = load_a_tile(hb, HID, m0, k0, lane);
                    v16bf b0 = load_b_tile(xlocW, HID, f0, k0, lane);
                    v16bf b1 = load_b_tile(xsclW, HID, f0, k0, lane);
                    cl = wmma_bf16(a, b0, cl);
                    cs = wmma_bf16(a, b1, cs);
                }
                float* dst = sRed + wib * 512 + (lane << 3);
                *(v4u*)(dst)       = ((const v4u*)&cl)[0];
                *(v4u*)(dst + 4)   = ((const v4u*)&cl)[1];
                *(v4u*)(dst + 256) = ((const v4u*)&cs)[0];
                *(v4u*)(dst + 260) = ((const v4u*)&cs)[1];
            }
            __syncthreads();
            if (blockIdx.x < 32 && wib == 0) {
                const int m0 = ((blockIdx.x >> 2) << 4);
                const int f0 = ((blockIdx.x & 3) << 4);
                const int fc = f0 + (lane & 15);
                const int mo = m0 + ((lane >> 4) << 3);
                for (int r = 0; r < 8; ++r) {
                    int e = (lane << 3) + r;
                    float yl = x_loc_b[fc];
                    float sp = x_scale_b[fc];
                    for (int w = 0; w < 8; ++w) {
                        yl += sRed[w * 512 + e];
                        sp += sRed[w * 512 + 256 + e];
                    }
                    int b = mo + r;
                    float ys = softplusf_(sp);
                    float xv = x[((size_t)b * SEQ + t) * NF + fc];
                    float dv = (xv - yl) / ys;
                    float lp = -0.5f * dv * dv - __logf(ys) - 0.5f * LOG2PI;
                    my_recon -= lp;
                    out[((size_t)b * SEQ + t) * NF + fc] = yl;   // y_loc output [b][t][f]
                }
            }
        }
        grid_barrier(bar);
    }

    // ---- deterministic final reductions for recon / kld ----
    if (blockIdx.x < 32 && (tid >> 5) == 0) red[blockIdx.x * 32 + lane] = my_recon;
    if ((gtid & 15) == 0) red[1024 + (gtid >> 4)] = my_kld;
    grid_barrier(bar);
    if (gtid == 0) {
        float rsum = 0.0f;
        for (int i = 0; i < 1024; ++i) rsum += red[i];
        float ksum = 0.0f;
        for (int i = 0; i < 2048; ++i) ksum += red[1024 + i];
        out[(size_t)BATCH * SEQ * NF + 0] = rsum;
        out[(size_t)BATCH * SEQ * NF + 1] = ksum;
    }
}

// ---------------- entry ----------------
extern "C" void kernel_launch(void* const* d_in, const int* in_sizes, int n_in,
                              void* d_out, int out_size, void* d_ws, size_t ws_size,
                              hipStream_t stream) {
    (void)in_sizes; (void)n_in; (void)out_size; (void)ws_size;
    const float* x           = (const float*)d_in[0];
    const float* eps         = (const float*)d_in[1];
    const float* phi_Wih     = (const float*)d_in[2];
    const float* phi_Whh     = (const float*)d_in[3];
    const float* phi_bih     = (const float*)d_in[4];
    const float* phi_bhh     = (const float*)d_in[5];
    const float* phi_loc_W   = (const float*)d_in[6];
    const float* phi_loc_b   = (const float*)d_in[7];
    const float* phi_scale_W = (const float*)d_in[8];
    const float* phi_scale_b = (const float*)d_in[9];
    const float* th_Wih      = (const float*)d_in[10];
    const float* th_Whh      = (const float*)d_in[11];
    const float* th_bih      = (const float*)d_in[12];
    const float* th_bhh      = (const float*)d_in[13];
    const float* x_loc_W     = (const float*)d_in[14];
    const float* x_loc_b     = (const float*)d_in[15];
    const float* x_scale_W   = (const float*)d_in[16];
    const float* x_scale_b   = (const float*)d_in[17];

    char*  ws  = (char*)d_ws;
    float* out = (float*)d_out;

    vrnn_prep<<<512, 256, 0, stream>>>(x, phi_Wih, phi_Whh, th_Wih, th_Whh,
                                       x_loc_W, x_scale_W, ws);
    vrnn_persist<<<NBLK, NTHR, 0, stream>>>(x, eps, phi_bih, phi_bhh,
                                            phi_loc_W, phi_loc_b,
                                            phi_scale_W, phi_scale_b,
                                            th_bih, th_bhh,
                                            x_loc_b, x_scale_b,
                                            ws, out);
}
